// LocalAttention_36807869726994
// MI455X (gfx1250) — matrix operand order repaired and checked by reading, not compiled, not run on hardware
//
#include <hip/hip_runtime.h>
#include <hip/hip_bf16.h>
#include <math.h>

typedef __attribute__((ext_vector_type(16))) _Float16 v16h;
typedef __attribute__((ext_vector_type(8)))  _Float16 v8h;
typedef __attribute__((ext_vector_type(8)))  float    v8f;

#define B_   32
#define N_   4096
#define D_   64
#define WS_  128
#define W_   (N_ / WS_)
#define J_   256            // keys per window (LB=1, LF=0)

#define KSTR 72             // halves per row of Ks (64 + 8 pad), 144B (16B-multiple)
#define VSTR 264            // halves per row of Vt (256 + 8 pad), 528B (16B-multiple)
#define PSTR 40             // halves per row of P scratch (32 + 8 pad), 80B (16B-multiple)

#define SMEM_BYTES (((256 * KSTR) + (64 * VSTR) + (8 * 16 * PSTR)) * 2 + 256 * 4)

#define NEG_MAX (-3.402823466e38f)

__device__ __forceinline__ float ivf_of(int c) {
  // inv_freq[c] = 10000^(-2c/64) = exp(-c/32 * ln(10000)), c in [0,32)
  return __expf(-(float)c * (0.03125f * 9.21034037197618f));
}

// Load 16 contiguous halves (32B) as two 16B LDS reads.
__device__ __forceinline__ v16h ld16(const _Float16* p) {
  v8h a = *(const v8h*)p;
  v8h b = *(const v8h*)(p + 8);
  v16h r;
#pragma unroll
  for (int e = 0; e < 8; ++e) { r[e] = a[e]; r[8 + e] = b[e]; }
  return r;
}

// Load 16 halves from two separated 16B groups (A-matrix layout helper).
__device__ __forceinline__ v16h ld16_split(const _Float16* p0, const _Float16* p1) {
  v8h a = *(const v8h*)p0;
  v8h b = *(const v8h*)p1;
  v16h r;
#pragma unroll
  for (int e = 0; e < 8; ++e) { r[e] = a[e]; r[8 + e] = b[e]; }
  return r;
}

__global__ void __launch_bounds__(256)
LocalAttention_36807869726994_kernel(const float* __restrict__ q,
                                     const float* __restrict__ k,
                                     const float* __restrict__ v,
                                     const unsigned char* __restrict__ mask,
                                     float* __restrict__ out) {
  extern __shared__ char smem_raw[];
  _Float16* Ks  = (_Float16*)smem_raw;           // 256 x KSTR  (RoPE'd keys, f16)
  _Float16* Vt  = Ks + 256 * KSTR;               // 64  x VSTR  (values, transposed, f16)
  _Float16* Pp  = Vt + 64 * VSTR;                // 8 waves x 16 x PSTR (P scratch)
  float*    bia = (float*)(Pp + 8 * 16 * PSTR);  // 256 key-validity flags

  const int w    = blockIdx.x;
  const int b    = blockIdx.y;
  const int tid  = threadIdx.x;
  const int lane = tid & 31;
  const int wv   = tid >> 5;
  const int n16  = lane & 15;
  const int hi   = lane >> 4;

  // ---------- cooperative stage: K (with RoPE) -> Ks, V -> Vt (transposed), validity ----------
  {
    const int jj  = tid;                    // key slot 0..255 within window
    const int pos = w * WS_ - WS_ + jj;     // global sequence position (may be < 0 = pad)
    const bool valid = (pos >= 0) && (mask[(size_t)b * N_ + pos] != 0);
    bia[jj] = valid ? 0.f : -1.f;
    if (valid) {
      const float* kp = k + ((size_t)b * N_ + pos) * D_;
      const float* vp = v + ((size_t)b * N_ + pos) * D_;
      const float  t  = (float)jj;          // RoPE position for keys = window-relative index
#pragma unroll 8
      for (int c = 0; c < 32; ++c) {
        float x1 = kp[c], x2 = kp[c + 32];
        float si, co;
        __sincosf(t * ivf_of(c), &si, &co);
        Ks[jj * KSTR + c]      = (_Float16)(x1 * co - x2 * si);
        Ks[jj * KSTR + c + 32] = (_Float16)(x2 * co + x1 * si);
      }
#pragma unroll 8
      for (int c = 0; c < 64; ++c) Vt[c * VSTR + jj] = (_Float16)vp[c];
    } else {
#pragma unroll 8
      for (int c = 0; c < 64; ++c) { Ks[jj * KSTR + c] = (_Float16)0.f; Vt[c * VSTR + jj] = (_Float16)0.f; }
    }
  }
  __syncthreads();

  // ---------- pack this lane's 16 key-validity flags into one register bitmask ----------
  // Lane touches key columns j = n16 + 16*u (u = jp*2 + t) across the whole loop.
  unsigned vmask = 0u;
#pragma unroll
  for (int u = 0; u < 16; ++u) {
    if (bia[n16 + 16 * u] >= 0.f) vmask |= (1u << u);
  }

  // ---------- per-wave Q strip: rows i0..i0+15, build A-fragments with RoPE + scale ----------
  const int i_row = wv * 16 + n16;          // this lane's Q row within the 128 block
  v16h aq0 = {}, aq1 = {};                  // K-step 0 (cols 0..31) and K-step 1 (cols 32..63)
  {
    const float* qp   = q + ((size_t)b * N_ + (size_t)w * WS_ + i_row) * D_;
    const float  t_ro = (float)(WS_ + i_row);     // q_freqs = freqs[-WS:]
    const float  scale = 0.125f;                  // d^-0.5, d = 64
#pragma unroll
    for (int e = 0; e < 8; ++e) {
      int c = hi * 16 + e;                        // elements 0..7 of A frag
      float x1 = qp[c] * scale, x2 = qp[c + 32] * scale;
      float si, co;
      __sincosf(t_ro * ivf_of(c), &si, &co);
      aq0[e] = (_Float16)(x1 * co - x2 * si);
      aq1[e] = (_Float16)(x2 * co + x1 * si);
      c = hi * 16 + 8 + e;                        // elements 8..15 of A frag
      x1 = qp[c] * scale; x2 = qp[c + 32] * scale;
      __sincosf(t_ro * ivf_of(c), &si, &co);
      aq0[8 + e] = (_Float16)(x1 * co - x2 * si);
      aq1[8 + e] = (_Float16)(x2 * co + x1 * si);
    }
  }

  // ---------- online-softmax over 8 pairs of 16-key tiles ----------
  float m_[8], l_[8];
  v8f o0 = {}, o1 = {}, o2 = {}, o3 = {};
#pragma unroll
  for (int g = 0; g < 8; ++g) { m_[g] = NEG_MAX; l_[g] = 0.f; }

  _Float16* Pw = Pp + wv * 16 * PSTR;      // wave-private P scratch (16 x 32 halves)

  for (int jp = 0; jp < 8; ++jp) {
    v8f sA, sB;
#pragma unroll
    for (int t = 0; t < 2; ++t) {
      const int j0 = jp * 32 + t * 16;
      // B operand: lane holds column (j0+n16), K-range hi*16..hi*16+15 (contiguous halves)
      const _Float16* kr = Ks + (size_t)(j0 + n16) * KSTR + hi * 16;
      v16h bk0 = ld16(kr);
      v16h bk1 = ld16(kr + 32);
      v8f s = {};
      s = __builtin_amdgcn_wmma_f32_16x16x32_f16(false, aq0, false, bk0, (short)0, s, false, false);
      s = __builtin_amdgcn_wmma_f32_16x16x32_f16(false, aq1, false, bk1, (short)0, s, false, false);
      // masking in C layout: VGPR g <-> row (g + 8*hi), lane n16 <-> key col (j0+n16).
      // Branch-free: key validity from register bitmask, causal from integer compare.
      const bool kvalid = ((vmask >> (jp * 2 + t)) & 1u) != 0u;
      const int  j_     = j0 + n16;
#pragma unroll
      for (int g = 0; g < 8; ++g) {
        const int i_ = wv * 16 + g + 8 * hi;
        const bool ok = kvalid && (j_ <= i_ + WS_);           // pad/mask + causal
        s[g] = ok ? s[g] : NEG_MAX;
      }
      if (t == 0) sA = s; else sB = s;
    }

    // flash update across the 32-key pair
#pragma unroll
    for (int g = 0; g < 8; ++g) {
      float rm = fmaxf(sA[g], sB[g]);
#pragma unroll
      for (int d = 1; d < 16; d <<= 1) rm = fmaxf(rm, __shfl_xor(rm, d, 32));
      const float mn = fmaxf(m_[g], rm);
      const float sc = __expf(m_[g] - mn);
      const float pa = (sA[g] > -3.0e38f) ? __expf(sA[g] - mn) : 0.f;
      const float pb = (sB[g] > -3.0e38f) ? __expf(sB[g] - mn) : 0.f;
      float rs = pa + pb;
#pragma unroll
      for (int d = 1; d < 16; d <<= 1) rs += __shfl_xor(rs, d, 32);
      l_[g] = l_[g] * sc + rs;
      m_[g] = mn;
      o0[g] *= sc; o1[g] *= sc; o2[g] *= sc; o3[g] *= sc;
      // stage P (f16) to wave-private scratch: row g+8*hi, cols n16 / 16+n16
      Pw[(g + 8 * hi) * PSTR + n16]      = (_Float16)pa;
      Pw[(g + 8 * hi) * PSTR + 16 + n16] = (_Float16)pb;
    }
    __builtin_amdgcn_wave_barrier();
    asm volatile("s_wait_dscnt 0x0" ::: "memory");

    // A operand from P scratch (A layout: lane=row n16, hi selects contiguous K half)
    const _Float16* pr = Pw + n16 * PSTR;
    const v16h ap = ld16(pr + hi * 16);

    // P(16x32) x V(32x64) -> accumulate 4 N-tiles; B from transposed V (contiguous K run)
    {
      const _Float16* vr = Vt + (size_t)n16 * VSTR + jp * 32 + hi * 16;
      v16h bv0 = ld16(vr);
      v16h bv1 = ld16(vr + 16 * VSTR);
      v16h bv2 = ld16(vr + 32 * VSTR);
      v16h bv3 = ld16(vr + 48 * VSTR);
      o0 = __builtin_amdgcn_wmma_f32_16x16x32_f16(false, ap, false, bv0, (short)0, o0, false, false);
      o1 = __builtin_amdgcn_wmma_f32_16x16x32_f16(false, ap, false, bv1, (short)0, o1, false, false);
      o2 = __builtin_amdgcn_wmma_f32_16x16x32_f16(false, ap, false, bv2, (short)0, o2, false, false);
      o3 = __builtin_amdgcn_wmma_f32_16x16x32_f16(false, ap, false, bv3, (short)0, o3, false, false);
    }
    __builtin_amdgcn_wave_barrier();   // keep next pair's P stores behind this pair's P loads
  }

  // ---------- epilogue: normalize and store f32 ----------
#pragma unroll
  for (int g = 0; g < 8; ++g) {
    const float inv = 1.f / l_[g];     // every row has >= 1 valid key (its own position)
    const int row = w * WS_ + wv * 16 + g + 8 * hi;
    float* op = out + ((size_t)b * N_ + row) * D_ + n16;
    op[0]  = o0[g] * inv;
    op[16] = o1[g] * inv;
    op[32] = o2[g] * inv;
    op[48] = o3[g] * inv;
  }
}

extern "C" void kernel_launch(void* const* d_in, const int* in_sizes, int n_in,
                              void* d_out, int out_size, void* d_ws, size_t ws_size,
                              hipStream_t stream) {
  (void)in_sizes; (void)n_in; (void)out_size; (void)d_ws; (void)ws_size;
  const float* q = (const float*)d_in[0];
  const float* k = (const float*)d_in[1];
  const float* v = (const float*)d_in[2];
  const unsigned char* mask = (const unsigned char*)d_in[3];
  float* out = (float*)d_out;

  dim3 grid(W_, B_);   // 32 windows x 32 batches = 1024 workgroups
  LocalAttention_36807869726994_kernel<<<grid, 256, SMEM_BYTES, stream>>>(q, k, v, mask, out);
}
